// MultiHeadSelfAttentionBlock_23330262351920
// MI455X (gfx1250) — matrix operand order repaired and checked by reading, not compiled, not run on hardware
//
#include <hip/hip_runtime.h>
#include <math.h>

// ---------------- problem constants ----------------
#define S    4096      // spatial (64*64)
#define C    256       // in channels
#define E    512       // embed
#define NH   8         // heads
#define DH   64        // head dim
#define EPSV 1e-5f

typedef _Float16 f16;
typedef _Float16 v16h __attribute__((ext_vector_type(16)));
typedef _Float16 v8h  __attribute__((ext_vector_type(8)));
typedef float    v8f  __attribute__((ext_vector_type(8)));

static __device__ __forceinline__ v8f wmma_f16(v16h a, v16h b, v8f c) {
  return __builtin_amdgcn_wmma_f32_16x16x32_f16(false, a, false, b, (short)0, c,
                                                false, false);
}

// A-matrix 16x32 f16 fragment: lane L -> row m=L&15, half hh=L>>4.
// VGPR0..3: K = hh*8 + {0..7}; VGPR4..7: K = 16 + hh*8 + {0..7}  (two b128 loads)
static __device__ __forceinline__ v16h load_afrag(const f16* row, int kbase, int hh) {
  union { v16h v; v8h h[2]; } u;
  u.h[0] = *(const v8h*)(row + kbase + hh * 8);
  u.h[1] = *(const v8h*)(row + kbase + 16 + hh * 8);
  return u.v;
}

// B-matrix 32x16 f16 fragment: lane L holds column n = col0 + (L&15); elements in
// table K order (s<8: K = hh*8+s; s>=8: K = 16+hh*8+(s-8)). p points at B[kbase][col0+(L&15)].
static __device__ __forceinline__ v16h load_bfrag(const f16* p, int ld, int hh) {
  v16h v;
#pragma unroll
  for (int s = 0; s < 8; ++s) {
    v[s] = p[(size_t)(hh * 8 + s) * ld];
    v[8 + s] = p[(size_t)(16 + hh * 8 + s) * ld];
  }
  return v;
}

// CDNA5 async cache->LDS DMA (ASYNCcnt tracked), 16 bytes per issue.
static __device__ __forceinline__ void async_to_lds_b128(const void* gptr, void* lds) {
  unsigned ldsaddr = (unsigned)(uintptr_t)lds;  // low 32 bits of generic = LDS addr
  asm volatile("global_load_async_to_lds_b128 %0, %1, off"
               :: "v"(ldsaddr), "v"(gptr)
               : "memory");
}
static __device__ __forceinline__ void wait_async0() {
  asm volatile("s_wait_asynccnt 0x0" ::: "memory");
}

// ---------------- kernel 1: bn1+relu, transpose to [S][C] f16 ----------------
__global__ void __launch_bounds__(256) k_prep_x(
    const float* __restrict__ x,
    const float* __restrict__ bw, const float* __restrict__ bb,
    const float* __restrict__ bm, const float* __restrict__ bv,
    f16* __restrict__ xa, f16* __restrict__ xr) {
  int gid = blockIdx.x * 256 + threadIdx.x;        // [0, C*S)
  int c = gid >> 12;                               // gid / S
  int s = gid & (S - 1);
  float xv = x[gid];                               // x is [c][s]
  float inv = bw[c] * rsqrtf(bv[c] + EPSV);
  float act = fmaxf(0.0f, xv * inv + (bb[c] - bm[c] * inv));
  xa[(size_t)s * C + c] = (f16)act;
  xr[(size_t)s * C + c] = (f16)xv;
}

// ---------------- kernel 2: transpose weights to f16 B layouts ----------------
__global__ void __launch_bounds__(256) k_prep_w(
    const float* __restrict__ Wq, const float* __restrict__ Wk,
    const float* __restrict__ Wv, const float* __restrict__ Wc,
    const float* __restrict__ Wr,
    f16* __restrict__ wT /*4x [C][E]*/, f16* __restrict__ wrT /*[E][C]*/) {
  int gid = blockIdx.x * 256 + threadIdx.x;        // [0, 5*E*C)
  const int EC = E * C;
  if (gid < 4 * EC) {
    int z = gid / EC;
    int j = gid - z * EC;                          // j = e*C + c  (src row-major (E,C))
    int e = j / C, c = j - e * C;
    const float* src = (z == 0) ? Wq : (z == 1) ? Wk : (z == 2) ? Wv : Wc;
    wT[(size_t)z * EC + (size_t)c * E + e] = (f16)src[j];
  } else {
    int j = gid - 4 * EC;                          // rs_w is (C,E): j = c*E + e
    int c = j / E, e = j - c * E;
    wrT[(size_t)e * C + c] = (f16)Wr[j];
  }
}

// ---------------- kernel 3: fused QKV + conv-path projection GEMMs -----------
__global__ void __launch_bounds__(32) k_proj(
    const f16* __restrict__ xa, const f16* __restrict__ xr,
    const f16* __restrict__ wT,
    const float* __restrict__ bq, const float* __restrict__ bk,
    const float* __restrict__ bvv, const float* __restrict__ cb,
    const float* __restrict__ cw, const float* __restrict__ cbb,
    const float* __restrict__ cm, const float* __restrict__ cv,
    const float* __restrict__ temp,
    f16* __restrict__ Qm, f16* __restrict__ Kt, f16* __restrict__ Vm,
    float* __restrict__ convp) {
  int s0 = blockIdx.x * 16, e0 = blockIdx.y * 16, z = blockIdx.z;
  int lane = threadIdx.x, m = lane & 15, hh = lane >> 4;
  const f16* A = (z == 3) ? xr : xa;
  const f16* W = wT + (size_t)z * C * E;
  const f16* arow = A + (size_t)(s0 + m) * C;
  v8f acc = {0.f, 0.f, 0.f, 0.f, 0.f, 0.f, 0.f, 0.f};
#pragma unroll
  for (int kk = 0; kk < C; kk += 32) {
    v16h a = load_afrag(arow, kk, hh);
    v16h b = load_bfrag(W + (size_t)kk * E + e0 + m, E, hh);
    acc = wmma_f16(a, b, acc);
  }
  int e = e0 + m;                                  // D col = lane&15
  int hd = e >> 6, d = e & 63;
  if (z == 0) {
    float it = 1.0f / temp[0];
    float bias = bq[e];
#pragma unroll
    for (int r = 0; r < 8; ++r) {
      int srow = s0 + hh * 8 + r;
      Qm[((size_t)hd * S + srow) * DH + d] = (f16)((acc[r] + bias) * it);
    }
  } else if (z == 1) {
    float bias = bk[e];
#pragma unroll
    for (int r = 0; r < 8; ++r) {
      int srow = s0 + hh * 8 + r;
      Kt[((size_t)hd * DH + d) * S + srow] = (f16)(acc[r] + bias);  // K^T
    }
  } else if (z == 2) {
    float bias = bvv[e];
#pragma unroll
    for (int r = 0; r < 8; ++r) {
      int srow = s0 + hh * 8 + r;
      Vm[((size_t)hd * S + srow) * DH + d] = (f16)(acc[r] + bias);
    }
  } else {
    float inv = cw[e] * rsqrtf(cv[e] + EPSV);
    float beta = cbb[e] - cm[e] * inv;
    float bias = cb[e];
#pragma unroll
    for (int r = 0; r < 8; ++r) {
      int srow = s0 + hh * 8 + r;
      float val = fmaxf(0.0f, (acc[r] + bias) * inv + beta);
      convp[(size_t)srow * E + e] = val;
    }
  }
}

// ---------------- kernel 4: flash attention + fuse with conv path ------------
__global__ void __launch_bounds__(128) k_attn(
    const f16* __restrict__ Qm, const f16* __restrict__ Kt,
    const f16* __restrict__ Vm, const float* __restrict__ convp,
    const float* __restrict__ gammap, f16* __restrict__ fused) {
  __shared__ f16 Ktile[64 * 64];     // [d][t]
  __shared__ f16 Vtile[64 * 64];     // [t][d]
  __shared__ f16 Ptile[4][16 * 64];  // per-wave probs [m][t]

  int h = blockIdx.y;
  int tid = threadIdx.x;
  int wave = tid >> 5, lane = tid & 31;
  int m = lane & 15, hh = lane >> 4;
  int q0 = blockIdx.x * 64 + wave * 16;

  const f16* qrow = Qm + ((size_t)h * S + q0 + m) * DH;
  v16h qa0 = load_afrag(qrow, 0, hh);
  v16h qa1 = load_afrag(qrow, 32, hh);

  const f16* Kbase = Kt + (size_t)h * DH * S;
  const f16* Vbase = Vm + (size_t)h * S * DH;

  float mrow[8], lrow[8];
  v8f accO[4];
#pragma unroll
  for (int r = 0; r < 8; ++r) { mrow[r] = -1e30f; lrow[r] = 0.0f; }
#pragma unroll
  for (int n = 0; n < 4; ++n) accO[n] = (v8f){0.f,0.f,0.f,0.f,0.f,0.f,0.f,0.f};

  for (int kb = 0; kb < S / 64; ++kb) {
    int t0 = kb * 64;
    // async DMA stage of K^T and V 64x64 f16 tiles directly into LDS
#pragma unroll
    for (int j = 0; j < 4; ++j) {
      int i = tid * 32 + j * 8;
      int row = i >> 6, col = i & 63;
      async_to_lds_b128(Kbase + (size_t)row * S + t0 + col, &Ktile[i]);
      async_to_lds_b128(Vbase + (size_t)(t0 + row) * DH + col, &Vtile[i]);
    }
    if (kb + 1 < S / 64) {  // prefetch next tiles -> global_prefetch_b8
      int row = tid >> 1, col = (tid & 1) * 32;
      __builtin_prefetch(Kbase + (size_t)row * S + (t0 + 64) + col, 0, 1);
      __builtin_prefetch(Vbase + (size_t)(t0 + 64 + row) * DH + col, 0, 1);
    }
    wait_async0();     // ASYNCcnt==0: this wave's DMA into LDS has landed
    __syncthreads();   // publish tiles across the 4 waves

    // scores: 16(q) x 64(k), 4 N-tiles x 2 K-steps
    v8f sc[4];
#pragma unroll
    for (int t = 0; t < 4; ++t) {
      v8f z = {0.f, 0.f, 0.f, 0.f, 0.f, 0.f, 0.f, 0.f};
      z = wmma_f16(qa0, load_bfrag(&Ktile[0 * 64 + t * 16 + m], 64, hh), z);
      z = wmma_f16(qa1, load_bfrag(&Ktile[32 * 64 + t * 16 + m], 64, hh), z);
      sc[t] = z;
    }

    // online softmax; rows (hh*8+r) live across the 16 lanes of each half-wave
    float alpha[8];
#pragma unroll
    for (int r = 0; r < 8; ++r) {
      float bmv = fmaxf(fmaxf(sc[0][r], sc[1][r]), fmaxf(sc[2][r], sc[3][r]));
      bmv = fmaxf(bmv, __shfl_xor(bmv, 1));
      bmv = fmaxf(bmv, __shfl_xor(bmv, 2));
      bmv = fmaxf(bmv, __shfl_xor(bmv, 4));
      bmv = fmaxf(bmv, __shfl_xor(bmv, 8));
      float nm = fmaxf(mrow[r], bmv);
      alpha[r] = __expf(mrow[r] - nm);
      mrow[r] = nm;
      float rs = 0.0f;
#pragma unroll
      for (int t = 0; t < 4; ++t) {
        float p = __expf(sc[t][r] - nm);
        sc[t][r] = p;
        rs += p;
      }
      rs += __shfl_xor(rs, 1);
      rs += __shfl_xor(rs, 2);
      rs += __shfl_xor(rs, 4);
      rs += __shfl_xor(rs, 8);
      lrow[r] = alpha[r] * lrow[r] + rs;
    }
#pragma unroll
    for (int n = 0; n < 4; ++n)
#pragma unroll
      for (int r = 0; r < 8; ++r) accO[n][r] *= alpha[r];

    // spill probs (f32 D-frag) to LDS as f16 A-layout [m][t]
    f16* pw = &Ptile[wave][0];
#pragma unroll
    for (int t = 0; t < 4; ++t)
#pragma unroll
      for (int r = 0; r < 8; ++r)
        pw[(hh * 8 + r) * 64 + t * 16 + m] = (f16)sc[t][r];

    const f16* prow = &Ptile[wave][m * 64];
    v16h pa0 = load_afrag(prow, 0, hh);
    v16h pa1 = load_afrag(prow, 32, hh);

    // out += P @ V : 4 d-tiles x 2 K-steps
#pragma unroll
    for (int n = 0; n < 4; ++n) {
      accO[n] = wmma_f16(pa0, load_bfrag(&Vtile[0 * 64 + n * 16 + m], 64, hh), accO[n]);
      accO[n] = wmma_f16(pa1, load_bfrag(&Vtile[32 * 64 + n * 16 + m], 64, hh), accO[n]);
    }
    __syncthreads();   // all waves done reading tiles before next DMA overwrite
  }

  float g = gammap[0];
#pragma unroll
  for (int n = 0; n < 4; ++n) {
    int e = h * DH + n * 16 + m;
#pragma unroll
    for (int r = 0; r < 8; ++r) {
      int srow = q0 + hh * 8 + r;
      float val = accO[n][r] / lrow[r];
      float fz = g * val + convp[(size_t)srow * E + e];
      fused[(size_t)srow * E + e] = (f16)fz;
    }
  }
}

// ---------------- kernel 5: residual 1x1 conv + bn + exact GELU -------------
__global__ void __launch_bounds__(32) k_final(
    const f16* __restrict__ fused, const f16* __restrict__ wrT,
    const float* __restrict__ rsb, const float* __restrict__ rw,
    const float* __restrict__ rb, const float* __restrict__ rm,
    const float* __restrict__ rv, float* __restrict__ out) {
  int s0 = blockIdx.x * 16, c0 = blockIdx.y * 16;
  int lane = threadIdx.x, m = lane & 15, hh = lane >> 4;
  const f16* arow = fused + (size_t)(s0 + m) * E;
  v8f acc = {0.f, 0.f, 0.f, 0.f, 0.f, 0.f, 0.f, 0.f};
#pragma unroll
  for (int kk = 0; kk < E; kk += 32) {
    v16h a = load_afrag(arow, kk, hh);
    v16h b = load_bfrag(wrT + (size_t)kk * C + c0 + m, C, hh);
    acc = wmma_f16(a, b, acc);
  }
  int c = c0 + m;
  float inv = rw[c] * rsqrtf(rv[c] + EPSV);
  float beta = rb[c] - rm[c] * inv;
  float bias = rsb[c];
#pragma unroll
  for (int r = 0; r < 8; ++r) {
    int srow = s0 + hh * 8 + r;
    float val = (acc[r] + bias) * inv + beta;
    val = 0.5f * val * (1.0f + erff(val * 0.70710678118654752f));
    out[(size_t)c * S + srow] = val;   // (B,C,H,W) layout, contiguous along s
  }
}

// ---------------- launcher ----------------
extern "C" void kernel_launch(void* const* d_in, const int* in_sizes, int n_in,
                              void* d_out, int out_size, void* d_ws, size_t ws_size,
                              hipStream_t stream) {
  const float* x     = (const float*)d_in[0];
  const float* bn1_w = (const float*)d_in[1];
  const float* bn1_b = (const float*)d_in[2];
  const float* bn1_m = (const float*)d_in[3];
  const float* bn1_v = (const float*)d_in[4];
  const float* Wq    = (const float*)d_in[5];
  const float* bq    = (const float*)d_in[6];
  const float* Wk    = (const float*)d_in[7];
  const float* bk    = (const float*)d_in[8];
  const float* Wv    = (const float*)d_in[9];
  const float* bvv   = (const float*)d_in[10];
  const float* gamma = (const float*)d_in[11];
  const float* temp  = (const float*)d_in[12];
  const float* conv_w = (const float*)d_in[13];
  const float* conv_b = (const float*)d_in[14];
  const float* cbn_w = (const float*)d_in[15];
  const float* cbn_b = (const float*)d_in[16];
  const float* cbn_m = (const float*)d_in[17];
  const float* cbn_v = (const float*)d_in[18];
  const float* rs_w  = (const float*)d_in[19];
  const float* rs_b  = (const float*)d_in[20];
  const float* rbn_w = (const float*)d_in[21];
  const float* rbn_b = (const float*)d_in[22];
  const float* rbn_m = (const float*)d_in[23];
  const float* rbn_v = (const float*)d_in[24];
  float* out = (float*)d_out;

  char* w = (char*)d_ws;
  size_t off = 0;
  auto take = [&](size_t bytes) { char* p = w + off; off += (bytes + 255) & ~(size_t)255; return p; };
  f16*   xa    = (f16*)take((size_t)S * C * 2);
  f16*   xr    = (f16*)take((size_t)S * C * 2);
  f16*   wT    = (f16*)take((size_t)4 * C * E * 2);
  f16*   wrT   = (f16*)take((size_t)E * C * 2);
  f16*   Qm    = (f16*)take((size_t)NH * S * DH * 2);
  f16*   Kt    = (f16*)take((size_t)NH * DH * S * 2);
  f16*   Vm    = (f16*)take((size_t)NH * S * DH * 2);
  float* convp = (float*)take((size_t)S * E * 4);
  f16*   fusedb= (f16*)take((size_t)S * E * 2);
  (void)ws_size; (void)in_sizes; (void)n_in; (void)out_size;

  k_prep_x<<<(C * S) / 256, 256, 0, stream>>>(x, bn1_w, bn1_b, bn1_m, bn1_v, xa, xr);
  k_prep_w<<<(5 * E * C) / 256, 256, 0, stream>>>(Wq, Wk, Wv, conv_w, rs_w, wT, wrT);
  k_proj<<<dim3(S / 16, E / 16, 4), 32, 0, stream>>>(
      xa, xr, wT, bq, bk, bvv, conv_b, cbn_w, cbn_b, cbn_m, cbn_v, temp,
      Qm, Kt, Vm, convp);
  k_attn<<<dim3(S / 64, NH), 128, 0, stream>>>(Qm, Kt, Vm, convp, gamma, fusedb);
  k_final<<<dim3(S / 16, C / 16), 32, 0, stream>>>(
      fusedb, wrT, rs_b, rbn_w, rbn_b, rbn_m, rbn_v, out);
}
